// SO2_Linear_28819230556353
// MI455X (gfx1250) — compile-verified, matrix operand order checked by reading, not run on hardware
//
#include <hip/hip_runtime.h>
#include <math.h>

// ---------------- types for WMMA ----------------
typedef __attribute__((ext_vector_type(16))) __bf16 bf16x16;
typedef __attribute__((ext_vector_type(8)))  float  f32x8;

struct __attribute__((aligned(16))) U128 { unsigned int a, b, c, d; };

// float -> bf16 (round to nearest even), stored as raw ushort
static __device__ __forceinline__ unsigned short f2bf(float f) {
    unsigned u = __builtin_bit_cast(unsigned, f);
    unsigned r = u + 0x7FFFu + ((u >> 16) & 1u);
    return (unsigned short)(r >> 16);
}

// Assemble a 16x32 (A) or 32x16 (B) bf16 fragment for one lane.
// Layout (CDNA5 ISA 7.12.2, 16-bit A 16x32): lane = {M|N} + 16*half,
// element e: K = kbase + (e>=8)*16 + half*8 + (e&7)  => two contiguous
// 8-element (16B) runs per lane.
static __device__ __forceinline__ bf16x16
ldfrag(const unsigned short* rowptr, int kbase, int half) {
    const unsigned short* p0 = rowptr + kbase + half * 8;
    U128 lo = *(const U128*)(p0);
    U128 hi = *(const U128*)(p0 + 16);
    union { U128 q[2]; bf16x16 v; } u;
    u.q[0] = lo; u.q[1] = hi;
    return u.v;
}

static __device__ __forceinline__ f32x8
wmma_bf16(bf16x16 a, bf16x16 b, f32x8 c) {
    return __builtin_amdgcn_wmma_f32_16x16x32_bf16(
        false, a, false, b, (short)0, c, false, false);
}

// Async global->LDS copy of one 16B chunk (ASYNCcnt-tracked).
// lds_off: LDS byte offset (low 32 bits of a generic pointer into __shared__).
static __device__ __forceinline__ void
async_cp16(unsigned lds_off, const void* gptr) {
    unsigned long long ga = (unsigned long long)(uintptr_t)gptr;
    asm volatile("global_load_async_to_lds_b128 %0, %1, off"
                 :: "v"(lds_off), "v"(ga) : "memory");
}
static __device__ __forceinline__ void async_wait0() {
    asm volatile("s_wait_asynccnt 0x0" ::: "memory");
}
static __device__ __forceinline__ unsigned lds_off_of(const void* p) {
    return (unsigned)(uintptr_t)p;   // LDS aperture: addr[31:0] = LDS offset
}

// ---------------- Wigner D matrices ----------------
static __device__ __forceinline__ void mm3(float* C, const float* A, const float* B) {
#pragma unroll
    for (int i = 0; i < 3; ++i)
#pragma unroll
        for (int j = 0; j < 3; ++j) {
            float s = 0.f;
#pragma unroll
            for (int k = 0; k < 3; ++k) s += A[i * 3 + k] * B[k * 3 + j];
            C[i * 3 + j] = s;
        }
}

static __device__ __forceinline__ void mm5(float* C, const float* A, const float* B) {
#pragma unroll
    for (int i = 0; i < 5; ++i)
#pragma unroll
        for (int j = 0; j < 5; ++j) {
            float s = 0.f;
#pragma unroll
            for (int k = 0; k < 5; ++k) s += A[i * 5 + k] * B[k * 5 + j];
            C[i * 5 + j] = s;
        }
}

static __device__ void compute_D(const float* Rv, float* D1, float* D2) {
    // v = R[[1,2,0]], normalized
    float vx = Rv[1], vy = Rv[2], vz = Rv[0];
    float nrm = sqrtf(vx * vx + vy * vy + vz * vz);
    nrm = fmaxf(nrm, 1e-12f);
    vx /= nrm; vy /= nrm; vz /= nrm;
    float beta  = acosf(fminf(fmaxf(vy, -1.f), 1.f));
    float alpha = atan2f(vx, vz);

    const float S3 = 0.8660254037844386f;
    const float J1[9]  = { 0,1,0,  1,0,0,  0,0,-1 };
    const float J2[25] = { 0,0,0,-1,0,
                           0,1,0,0,0,
                           0,0,-0.5f,0,-S3,
                           -1,0,0,0,0,
                           0,0,-S3,0,0.5f };

    float ca = cosf(alpha), sa = sinf(alpha);
    float cb = cosf(beta),  sb = sinf(beta);
    float Za1[9] = { ca,0,sa, 0,1,0, -sa,0,ca };
    float Zb1[9] = { cb,0,sb, 0,1,0, -sb,0,cb };
    float T1[9], T2[9];
    mm3(T1, Za1, J1); mm3(T2, T1, Zb1); mm3(D1, T2, J1);

    float c2a = cosf(2.f * alpha), s2a = sinf(2.f * alpha);
    float c2b = cosf(2.f * beta),  s2b = sinf(2.f * beta);
    float Za2[25] = { c2a,0,0,0,s2a,
                      0,ca,0,sa,0,
                      0,0,1,0,0,
                      0,-sa,0,ca,0,
                      -s2a,0,0,0,c2a };
    float Zb2[25] = { c2b,0,0,0,s2b,
                      0,cb,0,sb,0,
                      0,0,1,0,0,
                      0,-sb,0,cb,0,
                      -s2b,0,0,0,c2b };
    float U1[25], U2[25];
    mm5(U1, Za2, J2); mm5(U2, U1, Zb2); mm5(D2, U2, J2);
}

// ---------------- kernel 1: weights f32 -> bf16 ----------------
__global__ void prep_w(const float* __restrict__ W0, const float* __restrict__ W1,
                       const float* __restrict__ W2,
                       unsigned short* __restrict__ w0, unsigned short* __restrict__ w1,
                       unsigned short* __restrict__ w2) {
    const int N0 = 384 * 384, N1 = 512 * 256, N2 = 256 * 128;
    for (int i = blockIdx.x * blockDim.x + threadIdx.x; i < N0 + N1 + N2;
         i += gridDim.x * blockDim.x) {
        if (i < N0)            w0[i]           = f2bf(W0[i]);
        else if (i < N0 + N1)  w1[i - N0]      = f2bf(W1[i - N0]);
        else                   w2[i - N0 - N1] = f2bf(W2[i - N0 - N1]);
    }
}

// ---------------- kernel 2: forward rotation + gather into A0/A1/A2 ----------------
// A0: (n x 384)  m=0 input  (first 128 cols zero, l=0 x is unused)
// A1: (2n x 256) m=1 input  rows (e,c): c0 = l=1 pairs, c1 = l=2 pairs
// A2: (2n x 128) m=2 input  rows (e,c): flat IDX[2] split in halves
__global__ void rot_gather(const float* __restrict__ x, const float* __restrict__ R,
                           unsigned short* __restrict__ A0,
                           unsigned short* __restrict__ A1,
                           unsigned short* __restrict__ A2, int n) {
    int e = blockIdx.x;
    if (e >= n) return;
    int j = threadIdx.x;                       // mul index 0..127

    float D1[9], D2[25];
    compute_D(R + (size_t)e * 3, D1, D2);

    const float* xr = x + (size_t)e * 1152;
    float b1v[3], b2v[5];
#pragma unroll
    for (int k = 0; k < 3; ++k) b1v[k] = xr[128 + 3 * j + k];
#pragma unroll
    for (int k = 0; k < 5; ++k) b2v[k] = xr[512 + 5 * j + k];

    float r1[3], r2[5];
#pragma unroll
    for (int i = 0; i < 3; ++i) {              // r1 = b1 @ D1   (einsum 'nji,nmj->nmi')
        float s = 0.f;
#pragma unroll
        for (int k = 0; k < 3; ++k) s += D1[k * 3 + i] * b1v[k];
        r1[i] = s;
    }
#pragma unroll
    for (int i = 0; i < 5; ++i) {
        float s = 0.f;
#pragma unroll
        for (int k = 0; k < 5; ++k) s += D2[k * 5 + i] * b2v[k];
        r2[i] = s;
    }

    size_t a0 = (size_t)e * 384;
    A0[a0 + j]        = 0;                     // l=0 part zeroed per reference
    A0[a0 + 128 + j]  = f2bf(r1[1]);           // l=1 middle
    A0[a0 + 256 + j]  = f2bf(r2[2]);           // l=2 middle

    size_t a1r0 = ((size_t)e * 2 + 0) * 256;
    size_t a1r1 = ((size_t)e * 2 + 1) * 256;
    A1[a1r0 + 2 * j]     = f2bf(r1[0]);
    A1[a1r0 + 2 * j + 1] = f2bf(r1[2]);
    A1[a1r1 + 2 * j]     = f2bf(r2[1]);
    A1[a1r1 + 2 * j + 1] = f2bf(r2[3]);

    int p = 2 * j;                             // flat IDX[2] position
    size_t a2 = ((size_t)e * 2 + (p >> 7)) * 128 + (p & 127);
    A2[a2]     = f2bf(r2[0]);
    A2[a2 + 1] = f2bf(r2[4]);
}

// ---------------- kernel 3: m=0 GEMM  (n x 384) x W0^T + b0 -> scatter ----------------
// Block = 1 row tile (16 edges) x 4 column tiles; A tile staged in LDS via
// async copy, one wave per column tile.
__global__ void gemm_m0(const unsigned short* __restrict__ A0,
                        const unsigned short* __restrict__ W0bf,
                        const float* __restrict__ b0,
                        float* __restrict__ out, int n) {
    __shared__ __attribute__((aligned(16))) unsigned short sA[16 * 384]; // 12 KB

    int rt = blockIdx.x / 6;                // row tile
    int cg = blockIdx.x - rt * 6;           // column-tile group (4 tiles)
    if (rt >= (n >> 4)) return;
    int e0 = rt << 4;
    int lane = threadIdx.x & 31;
    int w    = threadIdx.x >> 5;
    int kc   = (cg * 4 + w) << 4;
    int m = lane & 15, half = lane >> 4;

    // ---- stage 16x384 bf16 A tile (contiguous in A0) into LDS ----
    const unsigned short* gsrc = A0 + (size_t)e0 * 384;
#pragma unroll
    for (int it = 0; it < 6; ++it) {        // 768 16B-chunks / 128 threads
        int i = it * 128 + threadIdx.x;
        async_cp16(lds_off_of(&sA[i * 8]), gsrc + i * 8);
    }
    async_wait0();
    __syncthreads();

    const unsigned short* arow = &sA[m * 384];
    const unsigned short* brow = W0bf + (size_t)(kc + m) * 384;  // row-major by out col

    f32x8 acc = {};
#pragma unroll
    for (int kb = 0; kb < 384; kb += 32) {
        __builtin_prefetch(brow + kb + 64, 0, 0);
        bf16x16 a = ldfrag(arow, kb, half);   // ds_load_b128 x2
        bf16x16 b = ldfrag(brow, kb, half);   // global_load_b128 x2 (L2-resident)
        acc = wmma_bf16(a, b, acc);
    }

    int ncol = kc + m;
    float bias = b0[ncol];
    int pos;
    if (ncol < 128)      pos = ncol;
    else if (ncol < 256) pos = 128 + (ncol - 128) * 3 + 1;
    else                 pos = 512 + (ncol - 256) * 5 + 2;
#pragma unroll
    for (int r = 0; r < 8; ++r) {
        int e = e0 + half * 8 + r;
        out[(size_t)e * 1152 + pos] = acc[r] + bias;
    }
}

// ---------------- kernel 4/5: m=1 / m=2 paired GEMM with +- combine ----------------
// Block = 1 row tile x 4 column tiles; 32xK A tile (rows e0*2 .. e0*2+31 are
// contiguous) staged in LDS; 4 accumulators per wave.
template <int MODE>
__global__ void gemm_pair(const unsigned short* __restrict__ A,
                          const unsigned short* __restrict__ Wbf,
                          float* __restrict__ out, int n) {
    constexpr int K  = (MODE == 1) ? 256 : 128;
    constexpr int NO = (MODE == 1) ? 256 : 128;   // half of output columns
    constexpr int CG = (MODE == 1) ? 4 : 2;       // column-tile groups of 4

    __shared__ __attribute__((aligned(16))) unsigned short sA[32 * K];  // 16/8 KB

    int rt = blockIdx.x / CG;
    int cg = blockIdx.x - rt * CG;
    if (rt >= (n >> 4)) return;
    int e0 = rt << 4;
    int lane = threadIdx.x & 31;
    int w    = threadIdx.x >> 5;
    int kc   = (cg * 4 + w) << 4;
    int m = lane & 15, half = lane >> 4;

    // ---- stage contiguous 32xK bf16 tile into LDS ----
    const unsigned short* gsrc = A + (size_t)e0 * 2 * K;
#pragma unroll
    for (int it = 0; it < K / 32; ++it) {     // (32*K/8) chunks / 128 threads
        int i = it * 128 + threadIdx.x;
        async_cp16(lds_off_of(&sA[i * 8]), gsrc + i * 8);
    }
    async_wait0();
    __syncthreads();

    const unsigned short* a0 = &sA[(m * 2 + 0) * K];
    const unsigned short* a1 = &sA[(m * 2 + 1) * K];
    const unsigned short* bl = Wbf + (size_t)(kc + m) * K;        // cols [kc, kc+16)
    const unsigned short* bh = Wbf + (size_t)(kc + NO + m) * K;   // cols [kc+NO, ...)

    f32x8 acc00 = {}, acc01 = {}, acc10 = {}, acc11 = {};
#pragma unroll
    for (int kb = 0; kb < K; kb += 32) {
        __builtin_prefetch(bl + kb + 64, 0, 0);
        __builtin_prefetch(bh + kb + 64, 0, 0);
        bf16x16 fa0 = ldfrag(a0, kb, half);
        bf16x16 fa1 = ldfrag(a1, kb, half);
        bf16x16 fb0 = ldfrag(bl, kb, half);
        bf16x16 fb1 = ldfrag(bh, kb, half);
        acc00 = wmma_bf16(fa0, fb0, acc00);
        acc01 = wmma_bf16(fa0, fb1, acc01);
        acc10 = wmma_bf16(fa1, fb0, acc10);
        acc11 = wmma_bf16(fa1, fb1, acc11);
    }

    int col = kc + m;
    int pre, pim;
    if (MODE == 1) {
        pre = 128 + 3 * (col >> 1) + ((col & 1) ? 2 : 0);   // IDX[1][col]
        pim = 512 + 5 * (col >> 1) + ((col & 1) ? 3 : 1);   // IDX[1][256+col]
    } else {
        int p = col;                                        // IDX[2][col]
        pre = 512 + 5 * (p >> 1) + ((p & 1) ? 4 : 0);
        int q = 128 + col;                                  // IDX[2][128+col]
        pim = 512 + 5 * (q >> 1) + ((q & 1) ? 4 : 0);
    }
#pragma unroll
    for (int r = 0; r < 8; ++r) {
        int e = e0 + half * 8 + r;
        float* orow = out + (size_t)e * 1152;
        orow[pre] = acc00[r] - acc11[r];   // out_re = y0[:no] - y1[no:]
        orow[pim] = acc01[r] + acc10[r];   // out_im = y0[no:] + y1[:no]
    }
}

// ---------------- kernel 6: inverse rotation, in place on d_out ----------------
__global__ void rot_out(const float* __restrict__ R, float* __restrict__ out, int n) {
    int e = blockIdx.x;
    if (e >= n) return;
    int j = threadIdx.x;

    float D1[9], D2[25];
    compute_D(R + (size_t)e * 3, D1, D2);

    float* orow = out + (size_t)e * 1152;
    float t1[3], t2[5];
#pragma unroll
    for (int k = 0; k < 3; ++k) t1[k] = orow[128 + 3 * j + k];
#pragma unroll
    for (int k = 0; k < 5; ++k) t2[k] = orow[512 + 5 * j + k];

    float o1[3], o2[5];
#pragma unroll
    for (int i = 0; i < 3; ++i) {              // einsum 'nij,nmj->nmi' : D @ t
        float s = 0.f;
#pragma unroll
        for (int k = 0; k < 3; ++k) s += D1[i * 3 + k] * t1[k];
        o1[i] = s;
    }
#pragma unroll
    for (int i = 0; i < 5; ++i) {
        float s = 0.f;
#pragma unroll
        for (int k = 0; k < 5; ++k) s += D2[i * 5 + k] * t2[k];
        o2[i] = s;
    }
#pragma unroll
    for (int k = 0; k < 3; ++k) orow[128 + 3 * j + k] = o1[k];
#pragma unroll
    for (int k = 0; k < 5; ++k) orow[512 + 5 * j + k] = o2[k];
}

// ---------------- launcher ----------------
extern "C" void kernel_launch(void* const* d_in, const int* in_sizes, int n_in,
                              void* d_out, int out_size, void* d_ws, size_t ws_size,
                              hipStream_t stream) {
    const float* x  = (const float*)d_in[0];
    const float* R  = (const float*)d_in[1];
    const float* W0 = (const float*)d_in[2];
    const float* b0 = (const float*)d_in[3];
    const float* W1 = (const float*)d_in[4];
    const float* W2 = (const float*)d_in[5];
    float* out = (float*)d_out;

    const int n = in_sizes[0] / 1152;          // 32768 edges

    // workspace layout (bf16 regions), all offsets 256B aligned
    char* ws = (char*)d_ws;
    unsigned short* w0bf = (unsigned short*)(ws);                       // 384*384
    unsigned short* w1bf = (unsigned short*)(ws + 294912);              // 512*256
    unsigned short* w2bf = (unsigned short*)(ws + 294912 + 262144);     // 256*128
    size_t oA0 = 294912 + 262144 + 65536;                               // = 622592
    unsigned short* A0 = (unsigned short*)(ws + oA0);                   // n*384
    unsigned short* A1 = (unsigned short*)(ws + oA0 + (size_t)n * 768); // 2n*256
    unsigned short* A2 = (unsigned short*)(ws + oA0 + (size_t)n * 768 + (size_t)n * 1024);

    // 1) weights -> bf16
    prep_w<<<608, 256, 0, stream>>>(W0, W1, W2, w0bf, w1bf, w2bf);

    // 2) forward rotation + gather
    rot_gather<<<n, 128, 0, stream>>>(x, R, A0, A1, A2, n);

    // 3) m=0 GEMM: one block per (row tile x 4 col tiles)
    int nrt = n >> 4;
    gemm_m0<<<nrt * 6, 128, 0, stream>>>(A0, w0bf, b0, out, n);

    // 4) m=1 paired GEMM
    gemm_pair<1><<<nrt * 4, 128, 0, stream>>>(A1, w1bf, out, n);

    // 5) m=2 paired GEMM
    gemm_pair<2><<<nrt * 2, 128, 0, stream>>>(A2, w2bf, out, n);

    // 6) inverse rotation in place
    rot_out<<<n, 128, 0, stream>>>(R, out, n);
}
